// SelfAttention_31155692765425
// MI455X (gfx1250) — matrix operand order repaired and checked by reading, not compiled, not run on hardware
//
#include <hip/hip_runtime.h>

#define B_ 8
#define S_ 2048
#define D_ 1024
#define SCALE 0.03125f   // 1/sqrt(1024)

typedef __bf16 bf16_t;
typedef __attribute__((ext_vector_type(16))) __bf16 v16bf;
typedef __attribute__((ext_vector_type(8)))  __bf16 v8bf;
typedef __attribute__((ext_vector_type(8)))  float  v8f;
typedef __attribute__((ext_vector_type(4)))  int    v4i;

union AB16 { v16bf v; v8bf h[2]; };

static __device__ inline v8f zero8() {
    v8f z = {0.f,0.f,0.f,0.f,0.f,0.f,0.f,0.f};
    return z;
}

// ---------------------------------------------------------------------------
// Async global->LDS copy (gfx1250 GLOBAL_LOAD_ASYNC_TO_LDS_B128, ASYNCcnt),
// with a synchronous fallback if the builtins are unavailable.
// Builtin signature (hipcc-verified): (v4i addrspace(1)*, v4i addrspace(3)*, imm, imm)
// ---------------------------------------------------------------------------
#if __has_builtin(__builtin_amdgcn_global_load_async_to_lds_b128) && \
    __has_builtin(__builtin_amdgcn_s_wait_asynccnt)
#define HAS_ASYNC_LDS 1
#else
#define HAS_ASYNC_LDS 0
#endif

typedef __attribute__((address_space(1))) v4i as1_v4i;
typedef __attribute__((address_space(3))) v4i as3_v4i;

// copy 16 bytes (8 x bf16) from global to LDS, per active lane
static __device__ inline void cp16_g2l(bf16_t* lds_dst, const bf16_t* gsrc) {
#if HAS_ASYNC_LDS
    __builtin_amdgcn_global_load_async_to_lds_b128((as1_v4i*)gsrc, (as3_v4i*)lds_dst, 0, 0);
#else
    *(v8bf*)lds_dst = *(const v8bf*)gsrc;
#endif
}

static __device__ inline void async_fence() {
#if HAS_ASYNC_LDS
    __builtin_amdgcn_s_wait_asynccnt(0);
#endif
}

// ---------------------------------------------------------------------------
// Fragment helpers (ISA 7.12.2 layouts)
// ---------------------------------------------------------------------------
// A-matrix fragment, 16x32 bf16 (MxK), from row-major bf16. Lane L: row=L&15,
// elements 0..7 at K = k0 + 8*(L>>4), elements 8..15 at +16.
static __device__ inline v16bf load_a_frag(const bf16_t* __restrict__ rowptr, int k0, int lane) {
    const int kb = (lane >> 4) << 3;
    AB16 u;
    u.h[0] = *(const v8bf*)(rowptr + k0 + kb);
    u.h[1] = *(const v8bf*)(rowptr + k0 + kb + 16);
    return u.v;
}

// Same A fragment sourced from f32 memory with on-the-fly bf16 conversion.
static __device__ inline v16bf load_a_frag_f32(const float* __restrict__ rowptr, int k0, int lane) {
    const int kb = (lane >> 4) << 3;
    v8f f0 = *(const v8f*)(rowptr + k0 + kb);
    v8f f1 = *(const v8f*)(rowptr + k0 + kb + 16);
    v16bf r;
#pragma unroll
    for (int i = 0; i < 8; ++i) { r[i] = (bf16_t)f0[i]; r[i + 8] = (bf16_t)f1[i]; }
    return r;
}

// B-matrix fragment, 32x16 bf16 (KxN), from an LDS tile laid out as
// [row][32 k-elements] where "row" is the N index (tile of 32 K per stage).
// Lane L: n = trow + (L&15); elements 0..7 at K = 8*(L>>4), elements 8..15 at +16.
static __device__ inline v16bf lds_b_frag(const bf16_t* ldsTile, int trow, int lane) {
    const int n_local = trow + (lane & 15);
    const int kb = (lane >> 4) << 3;
    AB16 u;
    u.h[0] = *(const v8bf*)(ldsTile + n_local * 32 + kb);
    u.h[1] = *(const v8bf*)(ldsTile + n_local * 32 + kb + 16);
    return u.v;
}

// Stage a [rows x 32] bf16 tile (rows*64 bytes) into LDS, 16B per chunk.
// chunk cid: row = cid>>2, part = cid&3.
static __device__ inline void stage_chunk(bf16_t* lds, const bf16_t* gbase, size_t rowStride,
                                          int k0, int cid) {
    const int row = cid >> 2, part = cid & 3;
    cp16_g2l(lds + row * 32 + part * 8, gbase + (size_t)row * rowStride + k0 + part * 8);
}

#define WMMA_BF16(a, b, c) \
    __builtin_amdgcn_wmma_f32_16x16x32_bf16(false, (a), false, (b), (short)0, (c), false, false)

// ---------------------------------------------------------------------------
// Kernel 1: f32 -> bf16 conversion (weights)
// ---------------------------------------------------------------------------
__global__ void cvt_f32_bf16_kernel(const float* __restrict__ src, bf16_t* __restrict__ dst, int n) {
    int i = blockIdx.x * blockDim.x + threadIdx.x;
    if (i < n) dst[i] = (bf16_t)src[i];
}

// ---------------------------------------------------------------------------
// Kernel 2: fused QKV projection. Block = 8 waves, tile 128 rows x 64 features.
// The three 64x32 weight tiles (12 KB) are async-staged into LDS, double
// buffered; each wave reads B-fragments from LDS in groups of 4 (batched
// ds waits), x A-fragments from global (converted f32->bf16 on the fly).
// V stored transposed: Vt[b][d][s].
// ---------------------------------------------------------------------------
__global__ __launch_bounds__(256) void qkv_proj_kernel(
    const float* __restrict__ x,
    const bf16_t* __restrict__ Wq, const bf16_t* __restrict__ Wk, const bf16_t* __restrict__ Wv,
    const float* __restrict__ bq, const float* __restrict__ bk, const float* __restrict__ bv,
    bf16_t* __restrict__ Q, bf16_t* __restrict__ K, bf16_t* __restrict__ Vt)
{
    __shared__ bf16_t wtile[2][3 * 64 * 32];   // 2 x 12 KB

    const int tid   = threadIdx.x;
    const int lane  = tid & 31;
    const int wave  = tid >> 5;
    const int rowBase = blockIdx.x * 128 + wave * 16;
    const int colBase = blockIdx.y * 64;
    const int b = rowBase / S_;
    const int nlane = lane & 15;
    const int rOff  = (lane >> 4) * 8;

    v8f accq[4], acck[4], accv[4];
#pragma unroll
    for (int t = 0; t < 4; ++t) { accq[t] = zero8(); acck[t] = zero8(); accv[t] = zero8(); }

    const float* xrow = x + (size_t)(rowBase + nlane) * D_;
    const bf16_t* Wbase[3] = { Wq + (size_t)colBase * D_,
                               Wk + (size_t)colBase * D_,
                               Wv + (size_t)colBase * D_ };

    // prologue: stage k-step 0 into buffer 0 (one 16B chunk per thread per matrix)
#pragma unroll
    for (int w = 0; w < 3; ++w) stage_chunk(&wtile[0][w * 64 * 32], Wbase[w], D_, 0, tid);

    for (int ki = 0; ki < D_ / 32; ++ki) {
        const int kt  = ki * 32;
        const int cur = ki & 1;
        async_fence();
        __syncthreads();
        if (ki + 1 < D_ / 32) {
#pragma unroll
            for (int w = 0; w < 3; ++w)
                stage_chunk(&wtile[cur ^ 1][w * 64 * 32], Wbase[w], D_, kt + 32, tid);
        }
        __builtin_prefetch(xrow + kt + 64, 0, 3);
        v16bf a = load_a_frag_f32(xrow, kt, lane);

        // group the 4 fragment loads per weight matrix before the 4 WMMAs
        v16bf fq[4], fk[4], fv[4];
#pragma unroll
        for (int t = 0; t < 4; ++t) fq[t] = lds_b_frag(&wtile[cur][0 * 64 * 32], t * 16, lane);
#pragma unroll
        for (int t = 0; t < 4; ++t) accq[t] = WMMA_BF16(a, fq[t], accq[t]);
#pragma unroll
        for (int t = 0; t < 4; ++t) fk[t] = lds_b_frag(&wtile[cur][1 * 64 * 32], t * 16, lane);
#pragma unroll
        for (int t = 0; t < 4; ++t) acck[t] = WMMA_BF16(a, fk[t], acck[t]);
#pragma unroll
        for (int t = 0; t < 4; ++t) fv[t] = lds_b_frag(&wtile[cur][2 * 64 * 32], t * 16, lane);
#pragma unroll
        for (int t = 0; t < 4; ++t) accv[t] = WMMA_BF16(a, fv[t], accv[t]);
    }

#pragma unroll
    for (int t = 0; t < 4; ++t) {
        const int n = colBase + t * 16 + nlane;
        const float biq = bq[n], bik = bk[n], biv = bv[n];
#pragma unroll
        for (int r = 0; r < 8; ++r) {
            const int row  = rowBase + rOff + r;
            const int srow = row - b * S_;
            Q[(size_t)row * D_ + n] = (bf16_t)(accq[t][r] + biq);
            K[(size_t)row * D_ + n] = (bf16_t)(acck[t][r] + bik);
            Vt[((size_t)b * D_ + n) * S_ + srow] = (bf16_t)(accv[t][r] + biv);
        }
    }
}

// ---------------------------------------------------------------------------
// Kernel 3: scores = Q K^T * SCALE with online row max/sum. Block = 8 waves,
// 128 query rows; keys processed in chunks of 128. The 128x32 K-tile (8 KB)
// is async-staged into LDS double buffered and shared by all 8 waves.
// ---------------------------------------------------------------------------
__global__ __launch_bounds__(256) void attn_scores_kernel(
    const bf16_t* __restrict__ Q, const bf16_t* __restrict__ K,
    bf16_t* __restrict__ Praw, float* __restrict__ ML)
{
    __shared__ bf16_t ktile[2][128 * 32];      // 2 x 8 KB

    const int tid  = threadIdx.x;
    const int lane = tid & 31;
    const int wave = tid >> 5;
    const int rowBase = blockIdx.x * 128 + wave * 16;
    const int b = rowBase / S_;
    const int nlane = lane & 15;
    const int rOff  = (lane >> 4) * 8;

    const bf16_t* qrow = Q + (size_t)(rowBase + nlane) * D_;

    float m[8], l[8];
#pragma unroll
    for (int r = 0; r < 8; ++r) { m[r] = -1e30f; l[r] = 0.f; }

    for (int c = 0; c < 16; ++c) {
        const int jBase = c * 128;
        const bf16_t* Kbase = K + (size_t)(b * S_ + jBase) * D_;

        v8f acc[8];
#pragma unroll
        for (int t = 0; t < 8; ++t) acc[t] = zero8();

        // prologue: two 16B chunks per thread fill the 8 KB tile
        stage_chunk(ktile[0], Kbase, D_, 0, tid);
        stage_chunk(ktile[0], Kbase, D_, 0, tid + 256);

        for (int ki = 0; ki < D_ / 32; ++ki) {
            const int kt  = ki * 32;
            const int cur = ki & 1;
            async_fence();
            __syncthreads();
            if (ki + 1 < D_ / 32) {
                stage_chunk(ktile[cur ^ 1], Kbase, D_, kt + 32, tid);
                stage_chunk(ktile[cur ^ 1], Kbase, D_, kt + 32, tid + 256);
            }
            v16bf a = load_a_frag(qrow, kt, lane);
            v16bf f[4];
#pragma unroll
            for (int t = 0; t < 4; ++t) f[t] = lds_b_frag(ktile[cur], t * 16, lane);
#pragma unroll
            for (int t = 0; t < 4; ++t) acc[t] = WMMA_BF16(a, f[t], acc[t]);
#pragma unroll
            for (int t = 0; t < 4; ++t) f[t] = lds_b_frag(ktile[cur], (t + 4) * 16, lane);
#pragma unroll
            for (int t = 0; t < 4; ++t) acc[t + 4] = WMMA_BF16(a, f[t], acc[t + 4]);
        }

        // online softmax stats (per row, reduced across 16 lanes of each half-wave)
#pragma unroll
        for (int r = 0; r < 8; ++r) {
            float cm = -1e30f;
#pragma unroll
            for (int t = 0; t < 8; ++t) cm = fmaxf(cm, acc[t][r]);
            cm *= SCALE;
#pragma unroll
            for (int off = 1; off <= 8; off <<= 1) cm = fmaxf(cm, __shfl_xor(cm, off, 32));
            const float nm = fmaxf(m[r], cm);
            float cs = 0.f;
#pragma unroll
            for (int t = 0; t < 8; ++t) cs += __expf(acc[t][r] * SCALE - nm);
#pragma unroll
            for (int off = 1; off <= 8; off <<= 1) cs += __shfl_xor(cs, off, 32);
            l[r] = l[r] * __expf(m[r] - nm) + cs;
            m[r] = nm;
        }

        // store raw scaled scores (bf16), coalesced along key dimension
#pragma unroll
        for (int t = 0; t < 8; ++t) {
#pragma unroll
            for (int r = 0; r < 8; ++r) {
                const int row = rowBase + rOff + r;
                const int col = jBase + t * 16 + nlane;
                Praw[(size_t)row * S_ + col] = (bf16_t)(acc[t][r] * SCALE);
            }
        }
    }

    if (nlane == 0) {
#pragma unroll
        for (int r = 0; r < 8; ++r) {
            const int row = rowBase + rOff + r;
            ML[row * 2 + 0] = m[r];
            ML[row * 2 + 1] = l[r];
        }
    }
}

// ---------------------------------------------------------------------------
// Kernel 4: O = softmax(P) @ V. Block = 8 waves as 4(M) x 2(N): 64 rows x
// 128 dims. The 128x32 V^T tile (8 KB) is async-staged into LDS double
// buffered; A-fragments apply exp(s-m)/l on the fly (TRANS co-executes with
// the XDL WMMA pipe).
// ---------------------------------------------------------------------------
__global__ __launch_bounds__(256) void attn_out_kernel(
    const bf16_t* __restrict__ Praw, const bf16_t* __restrict__ Vt,
    const float* __restrict__ ML, float* __restrict__ Out)
{
    __shared__ bf16_t vtile[2][128 * 32];      // 2 x 8 KB

    const int tid  = threadIdx.x;
    const int lane = tid & 31;
    const int wave = tid >> 5;
    const int wm = wave & 3, wn = wave >> 2;
    const int rowBase = blockIdx.x * 64 + wm * 16;
    const int colBase = blockIdx.y * 128;
    const int b = rowBase / S_;
    const int nlane = lane & 15;
    const int rOff  = (lane >> 4) * 8;

    const int arow = rowBase + nlane;
    const bf16_t* prow = Praw + (size_t)arow * S_;
    const float mrow = ML[arow * 2 + 0];
    const float invl = 1.0f / ML[arow * 2 + 1];

    const bf16_t* Vbase = Vt + ((size_t)b * D_ + colBase) * S_;

    v8f acc[4];
#pragma unroll
    for (int t = 0; t < 4; ++t) acc[t] = zero8();

    // prologue
    stage_chunk(vtile[0], Vbase, S_, 0, tid);
    stage_chunk(vtile[0], Vbase, S_, 0, tid + 256);

    for (int ji = 0; ji < S_ / 32; ++ji) {
        const int j   = ji * 32;
        const int cur = ji & 1;
        async_fence();
        __syncthreads();
        if (ji + 1 < S_ / 32) {
            stage_chunk(vtile[cur ^ 1], Vbase, S_, j + 32, tid);
            stage_chunk(vtile[cur ^ 1], Vbase, S_, j + 32, tid + 256);
        }
        __builtin_prefetch(prow + j + 64, 0, 3);
        const int kb = (lane >> 4) << 3;
        v8bf r0 = *(const v8bf*)(prow + j + kb);
        v8bf r1 = *(const v8bf*)(prow + j + kb + 16);
        v16bf f[4];
#pragma unroll
        for (int t = 0; t < 4; ++t) f[t] = lds_b_frag(vtile[cur], wn * 64 + t * 16, lane);
        v16bf a;
#pragma unroll
        for (int i = 0; i < 8; ++i) {
            a[i]     = (bf16_t)(__expf((float)r0[i] - mrow) * invl);
            a[i + 8] = (bf16_t)(__expf((float)r1[i] - mrow) * invl);
        }
#pragma unroll
        for (int t = 0; t < 4; ++t) acc[t] = WMMA_BF16(a, f[t], acc[t]);
    }

#pragma unroll
    for (int t = 0; t < 4; ++t) {
#pragma unroll
        for (int r = 0; r < 8; ++r) {
            const int row = rowBase + rOff + r;
            const int col = colBase + wn * 64 + t * 16 + nlane;
            Out[(size_t)row * D_ + col] = acc[t][r];
        }
    }
}

// ---------------------------------------------------------------------------
extern "C" void kernel_launch(void* const* d_in, const int* in_sizes, int n_in,
                              void* d_out, int out_size, void* d_ws, size_t ws_size,
                              hipStream_t stream) {
    const float* x  = (const float*)d_in[0];
    const float* Wq = (const float*)d_in[1];
    const float* bq = (const float*)d_in[2];
    const float* Wk = (const float*)d_in[3];
    const float* bk = (const float*)d_in[4];
    const float* Wv = (const float*)d_in[5];
    const float* bv = (const float*)d_in[6];
    float* out = (float*)d_out;

    // Workspace layout (bytes):
    //   Wq/Wk/Wv bf16 : 3 x 2 MB      @ 0, 2M, 4M
    //   Q, K          : 2 x 32 MB     @ 6M, 38M   (bf16, [B*S, D])
    //   Vt            : 32 MB         @ 70M       (bf16, [B, D, S])
    //   Praw          : 64 MB         @ 102M      (bf16, [B*S, S] raw scaled scores)
    //   ML            : 128 KB        @ 166M      (f32,  [B*S][m, l])
    const size_t MBy = 1024ull * 1024ull;
    char* ws = (char*)d_ws;
    bf16_t* wqb  = (bf16_t*)(ws + 0 * MBy);
    bf16_t* wkb  = (bf16_t*)(ws + 2 * MBy);
    bf16_t* wvb  = (bf16_t*)(ws + 4 * MBy);
    bf16_t* Qb   = (bf16_t*)(ws + 6 * MBy);
    bf16_t* Kb   = (bf16_t*)(ws + 38 * MBy);
    bf16_t* Vtb  = (bf16_t*)(ws + 70 * MBy);
    bf16_t* Praw = (bf16_t*)(ws + 102 * MBy);
    float*  ML   = (float*)(ws + 166 * MBy);

    {
        const int n = D_ * D_;
        dim3 grid(n / 256), block(256);
        cvt_f32_bf16_kernel<<<grid, block, 0, stream>>>(Wq, wqb, n);
        cvt_f32_bf16_kernel<<<grid, block, 0, stream>>>(Wk, wkb, n);
        cvt_f32_bf16_kernel<<<grid, block, 0, stream>>>(Wv, wvb, n);
    }
    {
        dim3 grid((B_ * S_) / 128, D_ / 64), block(256);
        qkv_proj_kernel<<<grid, block, 0, stream>>>(x, wqb, wkb, wvb, bq, bk, bv, Qb, Kb, Vtb);
    }
    {
        dim3 grid((B_ * S_) / 128), block(256);
        attn_scores_kernel<<<grid, block, 0, stream>>>(Qb, Kb, Praw, ML);
    }
    {
        dim3 grid((B_ * S_) / 64, D_ / 128), block(256);
        attn_out_kernel<<<grid, block, 0, stream>>>(Praw, Vtb, ML, out);
    }
}
